// GroupedQueryAttention_37074157699674
// MI455X (gfx1250) — compile-verified, matrix operand order checked by reading, not run on hardware
//
#include <hip/hip_runtime.h>
#include <hip/hip_bf16.h>

// ---------------------------------------------------------------------------
// MI455X / gfx1250 GQA attention, bf16 WMMA (v_wmma_f32_16x16x32_bf16), wave32
// All WMMA operands stored fragment-contiguous in LDS -> ds_load_b128 only.
// ---------------------------------------------------------------------------

typedef __attribute__((ext_vector_type(16))) __bf16 bf16x16;
typedef __attribute__((ext_vector_type(8)))  __bf16 bf16x8;
typedef __attribute__((ext_vector_type(4)))  __bf16 bf16x4;
typedef __attribute__((ext_vector_type(8)))  float  f32x8;

__device__ __forceinline__ f32x8 zero8() {
  f32x8 r;
#pragma unroll
  for (int i = 0; i < 8; ++i) r[i] = 0.0f;
  return r;
}

__device__ __forceinline__ bf16x4 cvt4(float a, float b, float c, float d) {
  bf16x4 r;
  r[0] = (__bf16)a; r[1] = (__bf16)b; r[2] = (__bf16)c; r[3] = (__bf16)d;
  return r;
}
__device__ __forceinline__ bf16x4 cvt4f(float4 f) { return cvt4(f.x, f.y, f.z, f.w); }

// WMMA operand fragment from a row-major [row][k] LDS tile (A-style; the B
// operand uses the same loader on a transposed [n][k] tile).
// Per ISA: lane_lo = row, half-wave hi selects K+8; elements K 0..7 and 16..23
// are contiguous per lane -> two 16-byte ds_load_b128.
__device__ __forceinline__ bf16x16 load_frag(const __bf16* smem, int stride,
                                             int row0, int k0, int lo, int hi) {
  const __bf16* p = smem + (row0 + lo) * stride + k0 + hi * 8;
  bf16x8 a0 = *(const bf16x8*)(p);
  bf16x8 a1 = *(const bf16x8*)(p + 16);
  return __builtin_shufflevector(a0, a1, 0, 1, 2, 3, 4, 5, 6, 7,
                                 8, 9, 10, 11, 12, 13, 14, 15);
}

#define WMMA_BF16(A, B, C)                                                    \
  __builtin_amdgcn_wmma_f32_16x16x32_bf16(false, (A), false, (B), (short)0,   \
                                          (C), false, false)

// ---------------------------------------------------------------------------
// GEMM: C = A(f32, MxK) @ B(f32, KxN), accum f32 via bf16 WMMA.
// EPI 0: plain row-major store  out[r*N + n]
// EPI 1: RoPE + scale, head-split store out[((b*H+h)*S+s)*128 + d]
// EPI 2: head-split store (no RoPE) — for V
// ---------------------------------------------------------------------------
#define GBM 128
#define GBN 128
#define GBK 32
#define STK (GBK + 8)   // 40 bf16 (80 B, 16B-aligned rows)
#define GSC (GBN + 4)   // 132 f32 (528 B rows)

template <int EPI>
__global__ __launch_bounds__(256) void gemm_bf16_wmma(
    const float* __restrict__ A, const float* __restrict__ Bw,
    float* __restrict__ out,
    const float* __restrict__ cosp, const float* __restrict__ sinp,
    int M, int N, int K, int H, int Sseq, float scale) {
  __shared__ __attribute__((aligned(16))) __bf16 As[GBM * STK];
  __shared__ __attribute__((aligned(16))) __bf16 Bt[GBN * STK];  // W transposed
  __shared__ __attribute__((aligned(16))) float  Cs[GBM * GSC];

  const int tid  = threadIdx.x;
  const int lane = tid & 31;
  const int w    = tid >> 5;   // 8 waves
  const int lo   = lane & 15;
  const int hi   = lane >> 4;
  const int wm   = w >> 1;     // 0..3 -> 32 rows each
  const int wn   = w & 1;      // 0..1 -> 64 cols each

  const int m0g = blockIdx.y * GBM;
  const int n0g = blockIdx.x * GBN;

  f32x8 acc[2][4];
#pragma unroll
  for (int mi = 0; mi < 2; ++mi)
#pragma unroll
    for (int ni = 0; ni < 4; ++ni) acc[mi][ni] = zero8();

  const int arow = tid >> 1;        // 0..127
  const int acol = (tid & 1) * 16;  // 0/16

  for (int kt = 0; kt < K; kt += GBK) {
    // A tile (128 x 32), row-major, bf16x4 stores (8B aligned)
    const float* ap = A + (size_t)(m0g + arow) * K + kt + acol;
#pragma unroll
    for (int v = 0; v < 4; ++v) {
      float4 f = *(const float4*)(ap + v * 4);
      *(bf16x4*)(As + arow * STK + acol + v * 4) = cvt4f(f);
    }
    // W tile transposed into Bt[n][k]: each thread owns k-quads; all four
    // global loads are lane-consecutive in n (coalesced), one b64 LDS store.
#pragma unroll
    for (int q = 0; q < 4; ++q) {
      int idx = tid + q * 256;
      int n   = idx & 127;
      int kq  = idx >> 7;  // 0..7
      const float* bp = Bw + (size_t)(kt + kq * 4) * N + n0g + n;
      *(bf16x4*)(Bt + n * STK + kq * 4) =
          cvt4(bp[0], bp[N], bp[2 * N], bp[3 * N]);
    }
    if (kt + GBK < K) {  // global_prefetch_b8 next tiles
      __builtin_prefetch(A + (size_t)(m0g + arow) * K + kt + GBK + acol, 0, 1);
      __builtin_prefetch(Bw + (size_t)(kt + GBK + (tid & 31)) * N + n0g +
                             (tid >> 5) * 16, 0, 1);
    }
    __syncthreads();

    bf16x16 afrag[2], bfrag[4];
#pragma unroll
    for (int mi = 0; mi < 2; ++mi)
      afrag[mi] = load_frag(As, STK, wm * 32 + mi * 16, 0, lo, hi);
#pragma unroll
    for (int ni = 0; ni < 4; ++ni)
      bfrag[ni] = load_frag(Bt, STK, wn * 64 + ni * 16, 0, lo, hi);
#pragma unroll
    for (int mi = 0; mi < 2; ++mi)
#pragma unroll
      for (int ni = 0; ni < 4; ++ni)
        acc[mi][ni] = WMMA_BF16(afrag[mi], bfrag[ni], acc[mi][ni]);
    __syncthreads();
  }

  // accumulators -> LDS (C/D layout: row = r + 8*hi, col = lane_lo)
#pragma unroll
  for (int mi = 0; mi < 2; ++mi)
#pragma unroll
    for (int ni = 0; ni < 4; ++ni)
#pragma unroll
      for (int r = 0; r < 8; ++r)
        Cs[(wm * 32 + mi * 16 + hi * 8 + r) * GSC + wn * 64 + ni * 16 + lo] =
            acc[mi][ni][r];
  __syncthreads();

  const int c    = tid & 63;
  const int r0   = tid >> 6;  // 0..3
  const int hloc = n0g >> 7;  // head index when N == H*128
  for (int rr = r0; rr < GBM; rr += 4) {
    int rg = m0g + rr;
    int bb = rg / Sseq;
    int ss = rg - bb * Sseq;
    if (EPI == 0) {
      out[(size_t)rg * N + n0g + c]      = Cs[rr * GSC + c];
      out[(size_t)rg * N + n0g + c + 64] = Cs[rr * GSC + c + 64];
    } else if (EPI == 1) {
      float x1 = Cs[rr * GSC + c];
      float x2 = Cs[rr * GSC + c + 64];
      float co = cosp[ss * 64 + c];
      float si = sinp[ss * 64 + c];
      float* op = out + (((size_t)bb * H + hloc) * Sseq + ss) * 128;
      op[c]      = (x1 * co - x2 * si) * scale;
      op[c + 64] = (x1 * si + x2 * co) * scale;
    } else {
      float* op = out + (((size_t)bb * H + hloc) * Sseq + ss) * 128;
      op[c]      = Cs[rr * GSC + c];
      op[c + 64] = Cs[rr * GSC + c + 64];
    }
  }
}

// ---------------------------------------------------------------------------
// Flash attention: per (b, h, 64-row q tile); causal; online softmax.
// Q pre-scaled by 1/sqrt(DH). K stored naturally (it IS the fragment-
// contiguous layout for Q@K^T); V stored transposed for P@V.
// ---------------------------------------------------------------------------
#define AQ  64
#define AK  128
#define ADH 128
#define ABS (AK + 8)    // 136 bf16: Qs/Ks/Vt/Ps row stride (272 B, 16B-aligned)
#define ASS (AK + 4)    // 132 f32

__global__ __launch_bounds__(256) void attn_wmma(
    const float* __restrict__ qws, const float* __restrict__ kc,
    const float* __restrict__ vc, float* __restrict__ ctx) {
  __shared__ __attribute__((aligned(16))) __bf16 Qs[AQ * ABS];    // [q][d]
  __shared__ __attribute__((aligned(16))) __bf16 Ks[AK * ABS];    // [kk][d]
  __shared__ __attribute__((aligned(16))) __bf16 Vt[ADH * ABS];   // [d][kk]
  __shared__ __attribute__((aligned(16))) __bf16 Ps[AQ * ABS];    // [q][kk]
  __shared__ __attribute__((aligned(16))) float  Ss[AQ * ASS];    // scores/O
  __shared__ float row_max[AQ], row_sum[AQ], row_scale[AQ];
  __shared__ float pmax[AQ][4], psum[AQ][4];

  const int tid  = threadIdx.x;
  const int lane = tid & 31;
  const int w    = tid >> 5;
  const int lo   = lane & 15;
  const int hi   = lane >> 4;
  const int wm   = w >> 2;  // 0..1 -> 32 rows each
  const int wn   = w & 3;   // 0..3 -> 32 cols each

  const int b   = blockIdx.z;
  const int h   = blockIdx.y;
  const int qi  = blockIdx.x;
  const int kvh = h >> 2;  // NG = NH/NKV = 4

  const float* qp = qws + ((size_t)b * 16 + h)   * 2048 * 128;
  const float* kp = kc  + ((size_t)b * 4  + kvh) * 2048 * 128;
  const float* vp = vc  + ((size_t)b * 4  + kvh) * 2048 * 128;

  // Q tile (rows qi*64 .. +63), f32 -> bf16, natural [q][d] layout
  {
    int r  = tid >> 2;
    int c0 = (tid & 3) * 32;
    const float* p = qp + (size_t)(qi * AQ + r) * 128 + c0;
#pragma unroll
    for (int v = 0; v < 8; ++v) {
      float4 f = *(const float4*)(p + v * 4);
      *(bf16x4*)(Qs + r * ABS + c0 + v * 4) = cvt4f(f);
    }
  }
  if (tid < AQ) { row_max[tid] = -3.0e38f; row_sum[tid] = 0.0f; }

  f32x8 oacc[2][2];
#pragma unroll
  for (int mi = 0; mi < 2; ++mi)
#pragma unroll
    for (int ni = 0; ni < 2; ++ni) oacc[mi][ni] = zero8();

  const int jlast = qi >> 1;  // causal coverage of this q tile
  for (int j = 0; j <= jlast; ++j) {
    __syncthreads();  // previous iteration done with Ks/Vt/Ss/Ps

    // K tile natural [kk][d]
    {
      int kk = tid >> 1;
      int d0 = (tid & 1) * 64;
      const float* p = kp + (size_t)(j * AK + kk) * 128 + d0;
#pragma unroll
      for (int v = 0; v < 16; ++v) {
        float4 f = *(const float4*)(p + v * 4);
        *(bf16x4*)(Ks + kk * ABS + d0 + v * 4) = cvt4f(f);
      }
    }
    // V tile transposed into Vt[d][kk] via kk-quads (coalesced in d)
#pragma unroll
    for (int q = 0; q < 16; ++q) {
      int idx = tid + q * 256;
      int d   = idx & 127;
      int kq  = idx >> 7;  // 0..31
      const float* p = vp + (size_t)(j * AK + kq * 4) * 128 + d;
      *(bf16x4*)(Vt + d * ABS + kq * 4) =
          cvt4(p[0], p[128], p[256], p[384]);
    }
    __syncthreads();

    // S = Q @ K^T (64 x 128); wave owns (wm*32 rows) x (wn*32 cols)
    f32x8 sacc[2][2];
#pragma unroll
    for (int mi = 0; mi < 2; ++mi)
#pragma unroll
      for (int ni = 0; ni < 2; ++ni) sacc[mi][ni] = zero8();
#pragma unroll
    for (int d0 = 0; d0 < ADH; d0 += 32) {
      bf16x16 a0 = load_frag(Qs, ABS, wm * 32 + 0,  d0, lo, hi);
      bf16x16 a1 = load_frag(Qs, ABS, wm * 32 + 16, d0, lo, hi);
      bf16x16 b0 = load_frag(Ks, ABS, wn * 32 + 0,  d0, lo, hi);
      bf16x16 b1 = load_frag(Ks, ABS, wn * 32 + 16, d0, lo, hi);
      sacc[0][0] = WMMA_BF16(a0, b0, sacc[0][0]);
      sacc[0][1] = WMMA_BF16(a0, b1, sacc[0][1]);
      sacc[1][0] = WMMA_BF16(a1, b0, sacc[1][0]);
      sacc[1][1] = WMMA_BF16(a1, b1, sacc[1][1]);
    }
#pragma unroll
    for (int mi = 0; mi < 2; ++mi)
#pragma unroll
      for (int ni = 0; ni < 2; ++ni)
#pragma unroll
        for (int r = 0; r < 8; ++r)
          Ss[(wm * 32 + mi * 16 + hi * 8 + r) * ASS + wn * 32 + ni * 16 + lo] =
              sacc[mi][ni][r];
    __syncthreads();

    // Online softmax: 4 threads per row, float4 reads, bf16x4 P stores.
    {
      const int r   = tid >> 2;
      const int sub = tid & 3;
      const float* srow = Ss + r * ASS + sub * 32;
      const int qg    = qi * AQ + r;
      const int kbase = j * AK + sub * 32;
      float mx = -3.0e38f;
#pragma unroll
      for (int c4 = 0; c4 < 32; c4 += 4) {
        float4 f = *(const float4*)(srow + c4);
        f.x = (kbase + c4 + 0 <= qg) ? f.x : -3.0e38f;
        f.y = (kbase + c4 + 1 <= qg) ? f.y : -3.0e38f;
        f.z = (kbase + c4 + 2 <= qg) ? f.z : -3.0e38f;
        f.w = (kbase + c4 + 3 <= qg) ? f.w : -3.0e38f;
        mx = fmaxf(mx, fmaxf(fmaxf(f.x, f.y), fmaxf(f.z, f.w)));
      }
      pmax[r][sub] = mx;
      __syncthreads();
      float m_old = row_max[r];
      float mnew  = fmaxf(m_old, fmaxf(fmaxf(pmax[r][0], pmax[r][1]),
                                       fmaxf(pmax[r][2], pmax[r][3])));
      __bf16* prow = Ps + r * ABS + sub * 32;
      float sum = 0.0f;
#pragma unroll
      for (int c4 = 0; c4 < 32; c4 += 4) {
        float4 f = *(const float4*)(srow + c4);
        float p0 = (kbase + c4 + 0 <= qg) ? __expf(f.x - mnew) : 0.0f;
        float p1 = (kbase + c4 + 1 <= qg) ? __expf(f.y - mnew) : 0.0f;
        float p2 = (kbase + c4 + 2 <= qg) ? __expf(f.z - mnew) : 0.0f;
        float p3 = (kbase + c4 + 3 <= qg) ? __expf(f.w - mnew) : 0.0f;
        sum += (p0 + p1) + (p2 + p3);
        *(bf16x4*)(prow + c4) = cvt4(p0, p1, p2, p3);
      }
      psum[r][sub] = sum;
      __syncthreads();
      if (sub == 0) {
        float scl    = __expf(m_old - mnew);
        row_scale[r] = scl;
        row_sum[r]   = row_sum[r] * scl +
                       (psum[r][0] + psum[r][1]) + (psum[r][2] + psum[r][3]);
        row_max[r]   = mnew;
      }
    }
    __syncthreads();

    // Rescale O by exp(m_old - m_new), then O += P @ V
#pragma unroll
    for (int mi = 0; mi < 2; ++mi)
#pragma unroll
      for (int r = 0; r < 8; ++r) {
        float scl = row_scale[wm * 32 + mi * 16 + hi * 8 + r];
#pragma unroll
        for (int ni = 0; ni < 2; ++ni) oacc[mi][ni][r] *= scl;
      }
#pragma unroll
    for (int k0 = 0; k0 < AK; k0 += 32) {
      bf16x16 a0 = load_frag(Ps, ABS, wm * 32 + 0,  k0, lo, hi);
      bf16x16 a1 = load_frag(Ps, ABS, wm * 32 + 16, k0, lo, hi);
      bf16x16 b0 = load_frag(Vt, ABS, wn * 32 + 0,  k0, lo, hi);
      bf16x16 b1 = load_frag(Vt, ABS, wn * 32 + 16, k0, lo, hi);
      oacc[0][0] = WMMA_BF16(a0, b0, oacc[0][0]);
      oacc[0][1] = WMMA_BF16(a0, b1, oacc[0][1]);
      oacc[1][0] = WMMA_BF16(a1, b0, oacc[1][0]);
      oacc[1][1] = WMMA_BF16(a1, b1, oacc[1][1]);
    }
  }

  __syncthreads();
  // O -> LDS, normalize by row_sum, write context as [b*S+s][h*128+d]
#pragma unroll
  for (int mi = 0; mi < 2; ++mi)
#pragma unroll
    for (int ni = 0; ni < 2; ++ni)
#pragma unroll
      for (int r = 0; r < 8; ++r)
        Ss[(wm * 32 + mi * 16 + hi * 8 + r) * ASS + wn * 32 + ni * 16 + lo] =
            oacc[mi][ni][r];
  __syncthreads();
  {
    int r  = tid >> 2;
    int c0 = (tid & 3) * 32;
    float inv = 1.0f / row_sum[r];
    float* op = ctx + ((size_t)b * 2048 + qi * AQ + r) * 2048 + h * 128 + c0;
#pragma unroll
    for (int c4 = 0; c4 < 32; c4 += 4) {
      float4 f = *(const float4*)(Ss + r * ASS + c0 + c4);
      f.x *= inv; f.y *= inv; f.z *= inv; f.w *= inv;
      *(float4*)(op + c4) = f;
    }
  }
}

// ---------------------------------------------------------------------------
// Launch: q-proj(+RoPE,scale) -> k-proj(+RoPE) -> v-proj -> flash attn -> o-proj
// d_out = [output (B*S*D)] [k-cache (B*NKV*S*DH)] [v-cache (B*NKV*S*DH)]
// d_ws  = [q rope'd (B*NH*S*DH)] [context (B*S*NH*DH)]  (~64 MB f32)
// ---------------------------------------------------------------------------
extern "C" void kernel_launch(void* const* d_in, const int* in_sizes, int n_in,
                              void* d_out, int out_size, void* d_ws,
                              size_t ws_size, hipStream_t stream) {
  const float* x    = (const float*)d_in[0];
  const float* cosp = (const float*)d_in[1];
  const float* sinp = (const float*)d_in[2];
  // d_in[3] = causal mask (bool) — recomputed from indices on device, ignored.
  const float* Wq   = (const float*)d_in[4];
  const float* Wk   = (const float*)d_in[5];
  const float* Wv   = (const float*)d_in[6];
  const float* Wo   = (const float*)d_in[7];

  const int B = 2, S = 2048, D = 2048, NH = 16, NKV = 4, DH = 128;
  float* out = (float*)d_out;
  const size_t outQ = (size_t)B * S * D;         // 8388608
  const size_t kvsz = (size_t)B * NKV * S * DH;  // 2097152
  float* kcache = out + outQ;
  float* vcache = out + outQ + kvsz;
  float* qws = (float*)d_ws;
  float* ctx = qws + (size_t)B * NH * S * DH;

  dim3 blk(256);
  const float qscale = 0.08838834764831845f;  // 1/sqrt(128)

  gemm_bf16_wmma<1><<<dim3((NH * DH) / GBN, (B * S) / GBM), blk, 0, stream>>>(
      x, Wq, qws, cosp, sinp, B * S, NH * DH, D, NH, S, qscale);
  gemm_bf16_wmma<1><<<dim3((NKV * DH) / GBN, (B * S) / GBM), blk, 0, stream>>>(
      x, Wk, kcache, cosp, sinp, B * S, NKV * DH, D, NKV, S, 1.0f);
  gemm_bf16_wmma<2><<<dim3((NKV * DH) / GBN, (B * S) / GBM), blk, 0, stream>>>(
      x, Wv, vcache, cosp, sinp, B * S, NKV * DH, D, NKV, S, 1.0f);
  attn_wmma<<<dim3(S / AQ, NH, B), blk, 0, stream>>>(qws, kcache, vcache, ctx);
  gemm_bf16_wmma<0><<<dim3(D / GBN, (B * S) / GBM), blk, 0, stream>>>(
      ctx, Wo, out, nullptr, nullptr, B * S, D, NH * DH, 0, S, 1.0f);
}
